// FusedExpertsNetwork_4939212391171
// MI455X (gfx1250) — compile-verified, matrix operand order checked by reading, not run on hardware
//
#include <hip/hip_runtime.h>
#include <hip/hip_bf16.h>
#include <stdint.h>

// ---------------------------------------------------------------------------
// Fused experts MLP:  y = relu(x @ W1^T + b1) @ W2 + b2   (batched over E)
// Two bf16-WMMA GEMM kernels, h chunked over H through d_ws (bf16).
// Register-pipelined staging: global prefetch of tile k+1 overlaps WMMA on k.
// ---------------------------------------------------------------------------

typedef __attribute__((ext_vector_type(16))) __bf16        v16bf;
typedef __attribute__((ext_vector_type(8)))  __bf16        v8bf;
typedef __attribute__((ext_vector_type(4)))  __bf16        bf16x4;
typedef __attribute__((ext_vector_type(8)))  float         v8f;
typedef __attribute__((ext_vector_type(4)))  float         f32x4;
typedef __attribute__((ext_vector_type(4)))  unsigned short u16x4;

#define E_ 8
#define C_ 4096
#define M_ 1024
#define H_ 4096
#define O_ 1024

#define BM 128          // block tile rows (C)
#define BN 128          // block tile cols (H-chunk cols / O cols)
#define BK 64           // block K step (2 WMMA k-slabs of 32)
#define LDSS 72         // LDS row stride in bf16 elems: 144 B = 9*16 B (aligned, staggered)
#define NTHREADS 256    // 8 wave32 waves: 2 (M) x 4 (N)
#define NSTG 8          // (BM*BK)/(4*NTHREADS) staging float4 per thread per tile

union V16U { v16bf v; v8bf h[2]; };

// Load one 16x32 bf16 A/B fragment from LDS in the CDNA5 WMMA lane layout:
//   lanes 0-15 : row = row0+lane,    K = {0..7, 16..23}  (+32*kstep)
//   lanes 16-31: row = row0+lane-16, K = {8..15, 24..31} (+32*kstep)
__device__ __forceinline__ v16bf load_frag(const unsigned short* base, int row0,
                                           int lane, int kstep) {
  int r  = row0 + (lane & 15);
  int kb = ((lane >> 4) << 3) + (kstep << 5);
  const unsigned short* p = base + r * LDSS + kb;
  V16U u;
  u.h[0] = *(const v8bf*)(p);        // 16 B aligned (row stride 144 B)
  u.h[1] = *(const v8bf*)(p + 16);
  return u.v;
}

// ---------------------------------------------------------------------------
// Kernel 1:  h[e, c, h0+n] = relu( sum_m x[e,c,m] * w1[e,h0+n,m] + b1[e,h0+n] )
// h stored bf16 in ws, layout [e][c][Hc] for the current chunk.
// ---------------------------------------------------------------------------
__global__ __launch_bounds__(NTHREADS)
void gemm1_relu_bf16(const float* __restrict__ x, const float* __restrict__ w1,
                     const float* __restrict__ b1, unsigned short* __restrict__ hws,
                     int h0, int Hc) {
  __shared__ unsigned short lds_a[BM * LDSS];
  __shared__ unsigned short lds_b[BN * LDSS];

  const int e       = blockIdx.y;
  const int tiles_n = Hc / BN;
  const int c0      = (blockIdx.x / tiles_n) * BM;
  const int n0      = (blockIdx.x % tiles_n) * BN;   // within chunk
  const int t       = threadIdx.x;
  const int lane    = t & 31;
  const int wave    = t >> 5;
  const int wm      = (wave & 1) * 64;   // wave tile: 64 x 32
  const int wn      = (wave >> 1) * 32;

  const int srow = t >> 4;          // staging: BK/4 = 16 float4 per row
  const int scol = (t & 15) * 4;    // 16 row-groups of 16 threads; +16 rows per it

  const float* Xb  = x  + ((size_t)e * C_ + c0 + srow) * M_ + scol;
  const float* W1b = w1 + ((size_t)e * H_ + h0 + n0 + srow) * M_ + scol;

  v8f acc[4][2] = {};
  f32x4 ra[NSTG], rb[NSTG];

  // prologue: load K-tile 0 into registers
#pragma unroll
  for (int it = 0; it < NSTG; ++it) {
    ra[it] = *(const f32x4*)(Xb  + (size_t)(it * 16) * M_);
    rb[it] = *(const f32x4*)(W1b + (size_t)(it * 16) * M_);
  }

  for (int k0 = 0; k0 < M_; k0 += BK) {
    // convert staged registers -> bf16 LDS tiles
#pragma unroll
    for (int it = 0; it < NSTG; ++it) {
      int off = (srow + it * 16) * LDSS + scol;
      *(bf16x4*)&lds_a[off] = __builtin_convertvector(ra[it], bf16x4);
      *(bf16x4*)&lds_b[off] = __builtin_convertvector(rb[it], bf16x4);
    }
    __syncthreads();

    // prefetch next K-tile (overlaps the WMMA block below)
    if (k0 + BK < M_) {
#pragma unroll
      for (int it = 0; it < NSTG; ++it) {
        ra[it] = *(const f32x4*)(Xb  + (size_t)(it * 16) * M_ + (k0 + BK));
        rb[it] = *(const f32x4*)(W1b + (size_t)(it * 16) * M_ + (k0 + BK));
      }
    }

#pragma unroll
    for (int ks = 0; ks < BK / 32; ++ks) {
      v16bf a[4], b[2];
#pragma unroll
      for (int fm = 0; fm < 4; ++fm) a[fm] = load_frag(lds_a, wm + fm * 16, lane, ks);
#pragma unroll
      for (int fn = 0; fn < 2; ++fn) b[fn] = load_frag(lds_b, wn + fn * 16, lane, ks);
#pragma unroll
      for (int fm = 0; fm < 4; ++fm)
#pragma unroll
        for (int fn = 0; fn < 2; ++fn)
          acc[fm][fn] = __builtin_amdgcn_wmma_f32_16x16x32_bf16(
              false, a[fm], false, b[fn], (short)0, acc[fm][fn], false, false);
    }
    __syncthreads();
  }

  // epilogue: bias + relu, store bf16 to ws
  float bias[2];
#pragma unroll
  for (int fn = 0; fn < 2; ++fn)
    bias[fn] = b1[(size_t)e * H_ + h0 + n0 + wn + fn * 16 + (lane & 15)];

#pragma unroll
  for (int fm = 0; fm < 4; ++fm)
#pragma unroll
    for (int fn = 0; fn < 2; ++fn)
#pragma unroll
      for (int i = 0; i < 8; ++i) {
        int crow = c0 + wm + fm * 16 + i + ((lane >> 4) << 3);
        int ncol = n0 + wn + fn * 16 + (lane & 15);
        float v  = fmaxf(acc[fm][fn][i] + bias[fn], 0.0f);
        *(__bf16*)&hws[((size_t)e * C_ + crow) * Hc + ncol] = (__bf16)v;
      }
}

// ---------------------------------------------------------------------------
// Kernel 2:  y[e,c,o] (+)= sum_{k<Hc} h[e,c,k] * w2[e,h0+k,o]   (+ b2 if first)
// ---------------------------------------------------------------------------
__global__ __launch_bounds__(NTHREADS)
void gemm2_acc_f32(const unsigned short* __restrict__ hws, const float* __restrict__ w2,
                   const float* __restrict__ b2, float* __restrict__ y,
                   int h0, int Hc, int first) {
  __shared__ unsigned short lds_a[BM * LDSS];
  __shared__ unsigned short lds_b[BN * LDSS];

  const int e    = blockIdx.y;
  const int c0   = (blockIdx.x / (O_ / BN)) * BM;
  const int n0   = (blockIdx.x % (O_ / BN)) * BN;   // O columns
  const int t    = threadIdx.x;
  const int lane = t & 31;
  const int wave = t >> 5;
  const int wm   = (wave & 1) * 64;
  const int wn   = (wave >> 1) * 32;

  // A staging: same row/col split as kernel 1, but bf16 source (u16x4)
  const int arow = t >> 4;
  const int acol = (t & 15) * 4;
  // B staging (transposed): BN/4 = 32 float4 per k-row; 8 k-rows per it-step
  const int bk   = t >> 5;          // k within tile (0..7), +8 per it
  const int bnv  = (t & 31) * 4;    // 4 consecutive O columns

  const unsigned short* Ab = hws + ((size_t)e * C_ + c0 + arow) * Hc + acol;
  const float* W2b = w2 + ((size_t)e * H_ + h0 + bk) * O_ + n0 + bnv;

  v8f acc[4][2] = {};
  u16x4 ra[NSTG];
  f32x4 rb[NSTG];

  // prologue: load K-tile 0
#pragma unroll
  for (int it = 0; it < NSTG; ++it) {
    ra[it] = *(const u16x4*)(Ab + (size_t)(it * 16) * Hc);
    rb[it] = *(const f32x4*)(W2b + (size_t)(it * 8) * O_);
  }

  for (int k0 = 0; k0 < Hc; k0 += BK) {
    // store staged tiles to LDS (A: copy, B: convert + transpose)
#pragma unroll
    for (int it = 0; it < NSTG; ++it) {
      *(u16x4*)&lds_a[(arow + it * 16) * LDSS + acol] = ra[it];
      bf16x4 c = __builtin_convertvector(rb[it], bf16x4);
      int k = bk + it * 8;
      *(__bf16*)&lds_b[(bnv + 0) * LDSS + k] = c[0];
      *(__bf16*)&lds_b[(bnv + 1) * LDSS + k] = c[1];
      *(__bf16*)&lds_b[(bnv + 2) * LDSS + k] = c[2];
      *(__bf16*)&lds_b[(bnv + 3) * LDSS + k] = c[3];
    }
    __syncthreads();

    // prefetch next K-tile (overlaps WMMA below)
    if (k0 + BK < Hc) {
#pragma unroll
      for (int it = 0; it < NSTG; ++it) {
        ra[it] = *(const u16x4*)(Ab + (size_t)(it * 16) * Hc + (k0 + BK));
        rb[it] = *(const f32x4*)(W2b + (size_t)(k0 + BK + it * 8) * O_);
      }
    }

#pragma unroll
    for (int ks = 0; ks < BK / 32; ++ks) {
      v16bf a[4], b[2];
#pragma unroll
      for (int fm = 0; fm < 4; ++fm) a[fm] = load_frag(lds_a, wm + fm * 16, lane, ks);
#pragma unroll
      for (int fn = 0; fn < 2; ++fn) b[fn] = load_frag(lds_b, wn + fn * 16, lane, ks);
#pragma unroll
      for (int fm = 0; fm < 4; ++fm)
#pragma unroll
        for (int fn = 0; fn < 2; ++fn)
          acc[fm][fn] = __builtin_amdgcn_wmma_f32_16x16x32_bf16(
              false, a[fm], false, b[fn], (short)0, acc[fm][fn], false, false);
    }
    __syncthreads();
  }

  // epilogue: accumulate into y (bias on first chunk, else read-modify-write)
  float bias[2];
#pragma unroll
  for (int fn = 0; fn < 2; ++fn)
    bias[fn] = first ? b2[(size_t)e * O_ + n0 + wn + fn * 16 + (lane & 15)] : 0.0f;

#pragma unroll
  for (int fm = 0; fm < 4; ++fm)
#pragma unroll
    for (int fn = 0; fn < 2; ++fn)
#pragma unroll
      for (int i = 0; i < 8; ++i) {
        int crow   = c0 + wm + fm * 16 + i + ((lane >> 4) << 3);
        int ocol   = n0 + wn + fn * 16 + (lane & 15);
        size_t idx = ((size_t)e * C_ + crow) * O_ + ocol;
        float v    = acc[fm][fn][i] + bias[fn];
        if (!first) v += y[idx];
        y[idx] = v;
      }
}

// ---------------------------------------------------------------------------
extern "C" void kernel_launch(void* const* d_in, const int* in_sizes, int n_in,
                              void* d_out, int out_size, void* d_ws, size_t ws_size,
                              hipStream_t stream) {
  (void)in_sizes; (void)n_in; (void)out_size;
  const float* x  = (const float*)d_in[0];
  const float* w1 = (const float*)d_in[1];
  const float* b1 = (const float*)d_in[2];
  const float* w2 = (const float*)d_in[3];
  const float* b2 = (const float*)d_in[4];
  float* y = (float*)d_out;
  unsigned short* hws = (unsigned short*)d_ws;

  // H-chunk size: largest {1024,512,256,128} whose bf16 h-chunk fits in ws
  int Hc = 128;
  if      ((size_t)E_ * C_ * 1024 * 2 <= ws_size) Hc = 1024;
  else if ((size_t)E_ * C_ *  512 * 2 <= ws_size) Hc = 512;
  else if ((size_t)E_ * C_ *  256 * 2 <= ws_size) Hc = 256;

  dim3 blk(NTHREADS);
  dim3 g1((C_ / BM) * (Hc / BN), E_);
  dim3 g2((C_ / BM) * (O_ / BN), E_);

  for (int h0 = 0; h0 < H_; h0 += Hc) {
    gemm1_relu_bf16<<<g1, blk, 0, stream>>>(x, w1, b1, hws, h0, Hc);
    gemm2_acc_f32 <<<g2, blk, 0, stream>>>(hws, w2, b2, y, h0, Hc, h0 == 0 ? 1 : 0);
  }
}
